// MultiHeadAttention_14903536517151
// MI455X (gfx1250) — matrix operand order repaired and checked by reading, not compiled, not run on hardware
//
#include <hip/hip_runtime.h>
#include <hip/hip_bf16.h>
#include <math.h>

typedef _Float16 half_t;
typedef __attribute__((ext_vector_type(16))) _Float16 v16h;
typedef __attribute__((ext_vector_type(8)))  float    v8f;

#define DEV __device__ __forceinline__

constexpr int Bsz  = 4;
constexpr int S    = 2048;
constexpr int DH   = 1024;    // == D_IN == D_OUT
constexpr int Mtot = Bsz * S; // 8192

// ---------------- WMMA fragment helpers (16x16x32 f16, wave32) ----------------
// A (16x32 f16): lane<16 row=lane, halves a[0..7]=K0..7, a[8..15]=K16..23
//                lane>=16 row=lane-16, a[0..7]=K8..15, a[8..15]=K24..31
DEV v16h load_a_f16(const half_t* base, int ld, int lane) {
  const half_t* p = base + (size_t)(lane & 15) * ld + ((lane >> 4) ? 8 : 0);
  v16h a;
#pragma unroll
  for (int i = 0; i < 8; ++i) a[i]     = p[i];
#pragma unroll
  for (int i = 0; i < 8; ++i) a[8 + i] = p[16 + i];
  return a;
}

// B (32x16 f16): lane<16 col=lane, b[0..7]=K0..7, b[8..15]=K16..23; lane>=16 col=lane-16, b[0..7]=K8..15, b[8..15]=K24..31
// With B = (row-major src)^T, each lane reads two runs of 8 contiguous f16 (16B) of one src row.
DEV v16h load_b_f16(const half_t* base, int ld, int lane) {
  const half_t* p = base + (size_t)(lane & 15) * ld + ((lane >> 4) ? 8 : 0);
  v16h b;
#pragma unroll
  for (int i = 0; i < 8; ++i) { b[i] = p[i]; b[8 + i] = p[16 + i]; }
  return b;
}

// Prefetch the line(s) a future load_b_f16 with the same (base, ld) will touch.
// Lowers to global_prefetch_b8 (WGP scope -> fills all cache levels; speculative,
// silently dropped on translation failure).
DEV void prefetch_b(const half_t* base, int ld, int lane) {
  const half_t* p = base + (size_t)(lane & 15) * ld + ((lane >> 4) ? 16 : 0);
  __builtin_prefetch((const void*)p, 0, 3);
}

DEV v8f wmma(v16h a, v16h b, v8f c) {
  return __builtin_amdgcn_wmma_f32_16x16x32_f16(false, a, false, b, (short)0, c,
                                                false, false);
}

// ---------------- f32 -> f16 weight pre-conversion (once per weight) ----------------
__global__ __launch_bounds__(256) void cvt_f32_f16(const float* __restrict__ src,
                                                   half_t* __restrict__ dst, int n) {
  int i = blockIdx.x * blockDim.x + threadIdx.x;
  const int stride = gridDim.x * blockDim.x;
  for (; i < n; i += stride) dst[i] = (_Float16)src[i];
}

// ---------------- GEMM: C[M,1024] = A[M,1024] @ Wh[1024,1024]^T + bias ----------------
// One workgroup (16 waves, 512 threads) owns one 16-row m-tile and all N=1024 cols.
// A tile (16x1024 f16 = 32KB) converted/staged in LDS once; each wave computes a
// 16x64 slab (4 WMMA tiles). Next k-step's weight lines are prefetched so the
// per-WMMA global loads hit near caches.
// OUT_MODE 0: f16 row-major   OUT_MODE 1: f16 transposed per batch [B][N][S] (for V)
// OUT_MODE 2: f32 row-major (final output)
template <bool A_F16, int OUT_MODE>
__global__ __launch_bounds__(512) void gemm_wmma(const void* __restrict__ Ap,
                                                 const half_t* __restrict__ Wh,
                                                 const float* __restrict__ bias,
                                                 void* __restrict__ Outp) {
  constexpr int K = 1024, N = 1024, NT = 4;
  const int lane = threadIdx.x & 31;
  const int wave = threadIdx.x >> 5; // 0..15
  const int m0 = blockIdx.x * 16;
  const int n0 = wave * (16 * NT); // 64 cols per wave

  __shared__ half_t sA[16 * K]; // 32 KB

  // cooperative stage: X[16,1024] -> f16 LDS (coalesced, one pass)
  if constexpr (A_F16) {
    const half_t* X = (const half_t*)Ap + (size_t)m0 * K;
    for (int i = threadIdx.x; i < 16 * K; i += 512) sA[i] = X[i];
  } else {
    const float* X = (const float*)Ap + (size_t)m0 * K;
    for (int i = threadIdx.x; i < 16 * K; i += 512) sA[i] = (_Float16)X[i];
  }
  // warm the first k-step's weight lines while waiting at the barrier
#pragma unroll
  for (int t = 0; t < NT; ++t) prefetch_b(Wh + (size_t)(n0 + t * 16) * K, K, lane);
  __syncthreads();

  v8f c[NT] = {};
  for (int k0 = 0; k0 < K; k0 += 32) {
    // prefetch next k-step's fragment lines (last-iter overshoot is speculative/harmless)
#pragma unroll
    for (int t = 0; t < NT; ++t)
      prefetch_b(Wh + (size_t)(n0 + t * 16) * K + k0 + 32, K, lane);
    v16h bf[NT];
#pragma unroll
    for (int t = 0; t < NT; ++t)
      bf[t] = load_b_f16(Wh + (size_t)(n0 + t * 16) * K + k0, K, lane);
    v16h a = load_a_f16(sA + k0, K, lane);
#pragma unroll
    for (int t = 0; t < NT; ++t) c[t] = wmma(a, bf[t], c[t]);
  }

  const int cl = lane & 15, ro = (lane >> 4) * 8;
#pragma unroll
  for (int t = 0; t < NT; ++t) {
    const int n = n0 + t * 16 + cl;
    const float bv = bias[n];
#pragma unroll
    for (int r = 0; r < 8; ++r) {
      const int m = m0 + ro + r;
      const float val = c[t][r] + bv;
      if constexpr (OUT_MODE == 0) {
        ((half_t*)Outp)[(size_t)m * N + n] = (_Float16)val;
      } else if constexpr (OUT_MODE == 1) {
        const int bb = m >> 11, ss = m & (S - 1);
        ((half_t*)Outp)[((size_t)bb * N + n) * S + ss] = (_Float16)val;
      } else {
        ((float*)Outp)[(size_t)m * N + n] = val;
      }
    }
  }
}

// ---------------- Flash attention ----------------
// Grid: (S/16, B). Block: 512 threads = 16 waves, 16 query rows per block.
// Waves split D=1024 into 16 slices of 64 cols for the PV accumulation.
// This key-block's V lines and next key-block's K lines are prefetched during the current block's
// softmax + barriers so fragment loads hit near caches.
__global__ __launch_bounds__(512) void attn_wmma(const half_t* __restrict__ Qp,
                                                 const half_t* __restrict__ Kp,
                                                 const half_t* __restrict__ Vt,
                                                 const int* __restrict__ mask,
                                                 half_t* __restrict__ Y) {
  constexpr int NW = 16, DSL = 64, NT = 4, KB = 32, NKB = S / KB;
  const int lane = threadIdx.x & 31;
  const int wave = threadIdx.x >> 5;
  const int b = blockIdx.y;
  const int m0 = blockIdx.x * 16;
  const int d0 = wave * DSL;

  const half_t* Q  = Qp + (size_t)b * S * DH;
  const half_t* Kb = Kp + (size_t)b * S * DH;
  const half_t* Vb = Vt + (size_t)b * DH * S; // V stored transposed [D][S]
  const int* maskB = mask + (size_t)b * S * S + (size_t)m0 * S;

  __shared__ float  sPart[NW][16][KB]; // 32 KB: per-wave partial scores
  __shared__ half_t sP[16][KB];        // 1 KB: softmax probs (WMMA A operand)
  __shared__ float  sScale[16];
  __shared__ float  sSum[16];

  // Q fragments for this wave's 64-d slice are loop-invariant: hoist.
  v16h aq[2];
#pragma unroll
  for (int j = 0; j < 2; ++j)
    aq[j] = load_a_f16(Q + (size_t)m0 * DH + d0 + 32 * j, DH, lane);

  v8f o[NT] = {};
  float m_run = -3.0e38f, l_run = 0.0f; // live only in wave 0

  const int cl = lane & 15, ro = (lane >> 4) * 8;

  // warm first key block's K lines
#pragma unroll
  for (int j = 0; j < 2; ++j) {
    prefetch_b(Kb + (size_t)0 * DH + d0 + 32 * j, DH, lane);
    prefetch_b(Kb + (size_t)16 * DH + d0 + 32 * j, DH, lane);
  }

  for (int kb = 0; kb < NKB; ++kb) {
    const int key0 = kb * KB;

    // ---- partial scores S_w[16,32]: batch all 4 K-fragment loads, then WMMAs
    v16h bk[4];
#pragma unroll
    for (int j = 0; j < 2; ++j) {
      bk[2 * j]     = load_b_f16(Kb + (size_t)key0 * DH + d0 + 32 * j, DH, lane);
      bk[2 * j + 1] = load_b_f16(Kb + (size_t)(key0 + 16) * DH + d0 + 32 * j, DH, lane);
    }
    // prefetch this block's V lines and next block's K lines; they stream in
    // while WMMAs + softmax + barriers run
#pragma unroll
    for (int t = 0; t < NT; ++t)
      prefetch_b(Vb + (size_t)(d0 + t * 16) * S + key0, S, lane);
    const int keyN = key0 + KB; // last-iter overshoot: speculative, harmless
#pragma unroll
    for (int j = 0; j < 2; ++j) {
      prefetch_b(Kb + (size_t)keyN * DH + d0 + 32 * j, DH, lane);
      prefetch_b(Kb + (size_t)(keyN + 16) * DH + d0 + 32 * j, DH, lane);
    }
    v8f s0 = {}, s1 = {};
#pragma unroll
    for (int j = 0; j < 2; ++j) {
      s0 = wmma(aq[j], bk[2 * j], s0);
      s1 = wmma(aq[j], bk[2 * j + 1], s1);
    }
#pragma unroll
    for (int r = 0; r < 8; ++r) {
      sPart[wave][ro + r][cl]      = s0[r];
      sPart[wave][ro + r][16 + cl] = s1[r];
    }
    __syncthreads();

    // ---- wave 0: reduce partials + masked online softmax (lane pairs share a row)
    if (wave == 0) {
      const int r  = lane & 15;
      const int j0 = (lane >> 4) * 16;
      float e[16];
#pragma unroll
      for (int j = 0; j < 16; ++j) e[j] = 0.0f;
#pragma unroll
      for (int w = 0; w < NW; ++w)
#pragma unroll
        for (int j = 0; j < 16; ++j) e[j] += sPart[w][r][j0 + j]; // 16B-contig: ds_load_b128
#pragma unroll
      for (int j = 0; j < 16; ++j) e[j] *= 0.03125f; // 1/sqrt(1024)
      const int* mrow = maskB + (size_t)r * S + key0 + j0;
#pragma unroll
      for (int j = 0; j < 16; ++j)
        if (mrow[j] == 0) e[j] = -32768.0f; // reference MASK_FILL = -2^15
      float bm = e[0];
#pragma unroll
      for (int j = 1; j < 16; ++j) bm = fmaxf(bm, e[j]);
      bm = fmaxf(bm, __shfl_xor(bm, 16, 32));
      const float mnew = fmaxf(m_run, bm);
      const float scl  = __expf(m_run - mnew);
      float psum = 0.0f;
#pragma unroll
      for (int j = 0; j < 16; ++j) {
        const float p = __expf(e[j] - mnew);
        psum += p;
        sP[r][j0 + j] = (_Float16)p;
      }
      psum += __shfl_xor(psum, 16, 32);
      l_run = l_run * scl + psum;
      m_run = mnew;
      if (j0 == 0) { sScale[r] = scl; sSum[r] = l_run; }
    }
    __syncthreads();

    // ---- rescale accumulators by per-row exp(m_old - m_new)
    float scl8[8];
#pragma unroll
    for (int r = 0; r < 8; ++r) scl8[r] = sScale[ro + r];
#pragma unroll
    for (int t = 0; t < NT; ++t)
#pragma unroll
      for (int r = 0; r < 8; ++r) o[t][r] *= scl8[r];

    // ---- PV: batch all 4 V-fragment loads, then WMMAs (A = probs, reused 4x)
    v16h ap = load_a_f16(&sP[0][0], KB, lane);
    v16h bv[NT];
#pragma unroll
    for (int t = 0; t < NT; ++t)
      bv[t] = load_b_f16(Vb + (size_t)(d0 + t * 16) * S + key0, S, lane);
#pragma unroll
    for (int t = 0; t < NT; ++t) o[t] = wmma(ap, bv[t], o[t]);

    __syncthreads(); // protect sPart/sP before next iteration overwrites
  }

  // ---- epilogue: divide by final row sums, store Y as f16 row-major
  float linv[8];
#pragma unroll
  for (int r = 0; r < 8; ++r) linv[r] = 1.0f / sSum[ro + r];
  half_t* Yb = Y + ((size_t)b * S + m0) * DH;
#pragma unroll
  for (int t = 0; t < NT; ++t)
#pragma unroll
    for (int r = 0; r < 8; ++r)
      Yb[(size_t)(ro + r) * DH + d0 + t * 16 + cl] = (_Float16)(o[t][r] * linv[r]);
}

// ---------------- launch ----------------
extern "C" void kernel_launch(void* const* d_in, const int* in_sizes, int n_in,
                              void* d_out, int out_size, void* d_ws, size_t ws_size,
                              hipStream_t stream) {
  (void)in_sizes; (void)n_in; (void)out_size; (void)ws_size;
  const float* q    = (const float*)d_in[0];
  const float* k    = (const float*)d_in[1];
  const float* v    = (const float*)d_in[2];
  const int*   mask = (const int*)d_in[3];
  const float* Wq   = (const float*)d_in[4];
  const float* bq   = (const float*)d_in[5];
  const float* Wk   = (const float*)d_in[6];
  const float* bk   = (const float*)d_in[7];
  const float* Wv   = (const float*)d_in[8];
  const float* bv   = (const float*)d_in[9];
  const float* Wo   = (const float*)d_in[10];
  const float* bo   = (const float*)d_in[11];

  char* ws = (char*)d_ws;
  const size_t szQKV = (size_t)Mtot * DH * sizeof(half_t); // 16 MB each
  const size_t szW   = (size_t)DH * DH * sizeof(half_t);   // 2 MB each
  half_t* Qp  = (half_t*)(ws);
  half_t* Kp  = (half_t*)(ws + szQKV);
  half_t* Vt  = (half_t*)(ws + 2 * szQKV);
  half_t* Y   = (half_t*)(ws + 3 * szQKV);
  half_t* WqH = (half_t*)(ws + 4 * szQKV);
  half_t* WkH = (half_t*)(ws + 4 * szQKV + szW);
  half_t* WvH = (half_t*)(ws + 4 * szQKV + 2 * szW);
  half_t* WoH = (half_t*)(ws + 4 * szQKV + 3 * szW);

  const int nW = DH * DH; // 1,048,576
  cvt_f32_f16<<<1024, 256, 0, stream>>>(Wq, WqH, nW);
  cvt_f32_f16<<<1024, 256, 0, stream>>>(Wk, WkH, nW);
  cvt_f32_f16<<<1024, 256, 0, stream>>>(Wv, WvH, nW);
  cvt_f32_f16<<<1024, 256, 0, stream>>>(Wo, WoH, nW);

  const int gemmBlocks = Mtot / 16; // 512: one 16-row m-tile per workgroup

  gemm_wmma<false, 0><<<gemmBlocks, 512, 0, stream>>>(q, WqH, bq, Qp);
  gemm_wmma<false, 0><<<gemmBlocks, 512, 0, stream>>>(k, WkH, bk, Kp);
  gemm_wmma<false, 1><<<gemmBlocks, 512, 0, stream>>>(v, WvH, bv, Vt); // transposed V
  attn_wmma<<<dim3(S / 16, Bsz), 512, 0, stream>>>(Qp, Kp, Vt, mask, Y);
  gemm_wmma<true, 2><<<gemmBlocks, 512, 0, stream>>>(Y, WoH, bo, d_out);
}
